// TDTransformer_74294344286257
// MI455X (gfx1250) — matrix operand order repaired and checked by reading, not compiled, not run on hardware
//
#include <hip/hip_runtime.h>
#include <hip/hip_bf16.h>
#include <stdint.h>

// ---------------------------------------------------------------------------
// MI455X (gfx1250): all GEMMs via v_wmma_f32_16x16x32_bf16.
//   conv GEMM : M=65536 K=1024 N=768   (im2col of k=2,s=2 patchify)
//   attention : per (b,h)/(b,w): 64x64x256 scores + 64x256x64 apply
//   proj GEMM : M=65536 K=512  N=256   (+bias+ReLU, coalesced fp32 NCHW out)
// A operands row-major [M][K]; B operands [N][K] -> all fragment loads are
// contiguous ds_load_b128.  q/k/v channels-last [b][h][w][c] bf16.
// Pure global->LDS copies use GLOBAL_LOAD_ASYNC_TO_LDS_B128 (ASYNCcnt path).
// ---------------------------------------------------------------------------

typedef __attribute__((ext_vector_type(16))) __bf16 bf16x16;
typedef __attribute__((ext_vector_type(8)))  __bf16 bf16x8;
typedef __attribute__((ext_vector_type(8)))  float  f32x8;
typedef __attribute__((__vector_size__(16))) int    i32x4;
typedef __bf16 bf16;

#define BLK 256   // 8 waves (wave32)

#if defined(__gfx1250__) && \
    __has_builtin(__builtin_amdgcn_global_load_async_to_lds_b128) && \
    __has_builtin(__builtin_amdgcn_s_wait_asynccnt)
#define HAS_ASYNC_LDS 1
#else
#define HAS_ASYNC_LDS 0
#endif

// 16-byte global->LDS copy: async DMA (ASYNCcnt) when available.
__device__ __forceinline__ void copy16_g2l(void* lds, const void* g) {
#if HAS_ASYNC_LDS
    __builtin_amdgcn_global_load_async_to_lds_b128(
        (__attribute__((address_space(1))) i32x4*)g,
        (__attribute__((address_space(3))) i32x4*)lds, 0, 0);
#else
    *(uint4*)lds = *(const uint4*)g;
#endif
}
__device__ __forceinline__ void copy_join() {
#if HAS_ASYNC_LDS
    __builtin_amdgcn_s_wait_asynccnt(0);
#endif
}

// ----- WMMA fragment loaders (wave32; A/B contiguous k = 16*(lane/16) + s) ----------

// A fragment 16x32 (MxK) from row-major [M][K]: per lane two contiguous 16B
// chunks: K = kb+[0,8) and kb+[8,16), kb = (lane<16?0:16).
__device__ __forceinline__ bf16x16 frag_a_rm(const bf16* p, int ld, int m0, int k0) {
    int l  = threadIdx.x & 31;
    const bf16* base = p + (m0 + (l & 15)) * ld + k0 + ((l < 16) ? 0 : 16);
    bf16x8 lo = *(const bf16x8*)base;
    bf16x8 hi = *(const bf16x8*)(base + 8);
    bf16x16 f;
#pragma unroll
    for (int i = 0; i < 8; ++i) { f[i] = lo[i]; f[i + 8] = hi[i]; }
    return f;
}

// B fragment 32x16 (KxN) from [N][K] storage: 16 contiguous bf16 per lane.
__device__ __forceinline__ bf16x16 frag_b_nk(const bf16* p, int ld, int k0, int n0) {
    int l = threadIdx.x & 31;
    const bf16* base = p + (n0 + (l & 15)) * ld + k0 + ((l < 16) ? 0 : 16);
    return *(const bf16x16*)base;
}

// C/D 16x16 f32: VGPR r -> (m = r + (lane<16?0:8), n = lane&15)
__device__ __forceinline__ void cd_coord(int r, int& dm, int& dn) {
    int l = threadIdx.x & 31;
    dm = r + ((l < 16) ? 0 : 8);
    dn = l & 15;
}

__device__ __forceinline__ f32x8 wmma_bf16(bf16x16 a, bf16x16 b, f32x8 c) {
    return __builtin_amdgcn_wmma_f32_16x16x32_bf16(false, a, false, b,
                                                   (short)0, c, false, false);
}

union Pack4 { bf16 h[4]; uint2 u; };
union Pack8 { bf16 h[8]; uint4 u; };

// ---------------------------------------------------------------------------
// Kernel 0: fold BN scales into weights ([N][K] layouts already, no transpose)
// ---------------------------------------------------------------------------
__global__ __launch_bounds__(BLK)
void prep_weights_kernel(const float* __restrict__ w_qkv,
                         const float* __restrict__ gamma1,
                         const float* __restrict__ w_proj,
                         const float* __restrict__ gamma2,
                         bf16* __restrict__ Bqkv, bf16* __restrict__ Bproj) {
    const float inv = rsqrtf(1.0f + 1e-5f);
    int idx = blockIdx.x * BLK + threadIdx.x;
    const int N1 = 768 * 1024;
    if (idx < N1) {
        int n = idx >> 10;
        Bqkv[idx] = (bf16)(w_qkv[idx] * (gamma1[n] * inv));
    } else {
        int j = idx - N1;
        if (j < 256 * 512) {
            int n = j >> 9;
            Bproj[j] = (bf16)(w_proj[j] * (gamma2[n] * inv));
        }
    }
}

// ---------------------------------------------------------------------------
// Kernel 1: patchify conv GEMM.  WG tile 128x64, wave tile 32x32 (2x2 WMMA).
// im2col: one float4 covers the 2x2-patch k-quad of TWO adjacent output
// columns (rows m, m+1) -> b128 loads, ds_store_b64 per row.
// Epilogue: +beta1, channels-last q/k/v.
// ---------------------------------------------------------------------------
__global__ __launch_bounds__(BLK)
void conv_qkv_kernel(const float* __restrict__ x,
                     const bf16* __restrict__ Bqkv,
                     const float* __restrict__ beta1,
                     bf16* __restrict__ qT, bf16* __restrict__ kT,
                     bf16* __restrict__ vT) {
    extern __shared__ __align__(32) char smem[];
    bf16* As = (bf16*)smem;                    // [128][32]  (M x K)
    bf16* Bs = (bf16*)(smem + 128 * 32 * 2);   // [64][32]   (N x K)

    const int mTile = blockIdx.x * 128;
    const int nTile = blockIdx.y * 64;
    const int t     = threadIdx.x;
    const int wave  = t >> 5;
    const int wm    = (wave & 3) * 32;
    const int wn    = (wave >> 2) * 32;

    f32x8 acc[2][2] = {{{}, {}}, {{}, {}}};

    for (int k0 = 0; k0 < 1024; k0 += 32) {
        const int ci0 = k0 >> 2;               // 8 input channels per K-step
        // ---- stage B first (async path overlaps with A gather)
        {
            int n = t >> 2, q = t & 3;
            copy16_g2l(&Bs[n * 32 + q * 8],
                       &Bqkv[(size_t)(nTile + n) * 1024 + k0 + q * 8]);
        }
        // ---- stage A: 64 row-pairs x 8 ci, 2 tasks/thread
        float4 y0[2], y1[2];
#pragma unroll
        for (int j = 0; j < 2; ++j) {
            int f = t + j * BLK;
            int pairIdx = f >> 3, cidx = f & 7;
            int m = mTile + pairIdx * 2;
            int b = m >> 12, oy = (m >> 6) & 63, ox = m & 63;
            const float* src =
                x + (((b * 256 + ci0 + cidx) * 128) + oy * 2) * 128 + ox * 2;
            y0[j] = *(const float4*)src;          // ky=0: rows m,m+1
            y1[j] = *(const float4*)(src + 128);  // ky=1: rows m,m+1
        }
#pragma unroll
        for (int j = 0; j < 2; ++j) {
            int f = t + j * BLK;
            int rowL = (f >> 3) * 2, cidx = f & 7;
            Pack4 p0, p1;
            p0.h[0] = (bf16)y0[j].x; p0.h[1] = (bf16)y0[j].y;
            p0.h[2] = (bf16)y1[j].x; p0.h[3] = (bf16)y1[j].y;
            p1.h[0] = (bf16)y0[j].z; p1.h[1] = (bf16)y0[j].w;
            p1.h[2] = (bf16)y1[j].z; p1.h[3] = (bf16)y1[j].w;
            *(uint2*)&As[rowL * 32 + cidx * 4]       = p0.u;
            *(uint2*)&As[(rowL + 1) * 32 + cidx * 4] = p1.u;
        }
        if (k0 + 32 < 1024)  // global_prefetch for next B block
            __builtin_prefetch(&Bqkv[(size_t)(nTile + (t >> 2)) * 1024 + k0 + 32], 0, 1);
        copy_join();
        __syncthreads();
#pragma unroll
        for (int i = 0; i < 2; ++i) {
            bf16x16 a = frag_a_rm(As, 32, wm + i * 16, 0);
#pragma unroll
            for (int jj = 0; jj < 2; ++jj) {
                bf16x16 b = frag_b_nk(Bs, 32, 0, wn + jj * 16);
                acc[i][jj] = wmma_bf16(a, b, acc[i][jj]);
            }
        }
        __syncthreads();
    }

    // epilogue: +beta1, write channels-last q/k/v
#pragma unroll
    for (int i = 0; i < 2; ++i)
#pragma unroll
        for (int jj = 0; jj < 2; ++jj) {
            int nBase = nTile + wn + jj * 16;
            bf16* dst = (nBase < 256) ? qT : (nBase < 512) ? kT : vT;
            int cBase = nBase & 255;
#pragma unroll
            for (int r = 0; r < 8; ++r) {
                int dm, dn; cd_coord(r, dm, dn);
                int m = mTile + wm + i * 16 + dm;
                float y = acc[i][jj][r] + beta1[nBase + dn];
                int b = m >> 12, oy = (m >> 6) & 63, ox = m & 63;
                dst[((size_t)((b * 64 + oy) * 64 + ox)) * 256 + cBase + dn] = (bf16)y;
            }
        }
}

// ---------------------------------------------------------------------------
// Kernel 2: height-axis attention.  One WG per (b,h).
//   S[w][v] = sum_c Ks[w][c]*Qs[v][c] /16 ; softmax over w (per column v)
//   Out[v][c] = sum_w Pt[v][w]*Vt[c][w]   -> vcat[b][h][v][0:256]
// ---------------------------------------------------------------------------
__global__ __launch_bounds__(BLK)
void attn_h_kernel(const bf16* __restrict__ qT, const bf16* __restrict__ kT,
                   const bf16* __restrict__ vT, bf16* __restrict__ vcat) {
    extern __shared__ __align__(32) char smem[];
    bf16*  Ks = (bf16*)smem;              // [64][256]  A: [w][c]
    bf16*  Qs = Ks + 64 * 256;            // [64][256]  B(NK): [v][c]
    bf16*  Vt = Qs + 64 * 256;            // [256][64]  B(NK): [c][w]
    float* S  = (float*)(Vt + 256 * 64);  // [64][64]   scores [w][v]
    bf16*  Pt = (bf16*)(S + 64 * 64);     // [64][64]   A: [v][w]

    const int blk = blockIdx.x;
    const int b = blk >> 6, hh = blk & 63;
    const int t = threadIdx.x, wave = t >> 5;
    const size_t sliceOff = (size_t)(b * 64 + hh) * 64 * 256;

    // ---- stage K,Q: 4096 uint4 each (async DMA path)
    {
        const uint4* sk = (const uint4*)(kT + sliceOff);
        const uint4* sq = (const uint4*)(qT + sliceOff);
        uint4* dk = (uint4*)Ks; uint4* dq = (uint4*)Qs;
#pragma unroll
        for (int j = 0; j < 16; ++j) {
            int f = t + j * BLK;
            copy16_g2l(&dk[f], &sk[f]);
            copy16_g2l(&dq[f], &sq[f]);
        }
    }
    // ---- stage V transposed: read [w][c] as uint4 (8 ch), scatter to Vt[c][w]
    {
        const uint4* sv = (const uint4*)(vT + sliceOff);
#pragma unroll
        for (int j = 0; j < 8; ++j) {
            int f = t + j * BLK;          // 2048 chunks
            int w = f >> 5, chunk = f & 31;
            Pack8 p; p.u = sv[f];
#pragma unroll
            for (int i = 0; i < 8; ++i)
                Vt[(chunk * 8 + i) * 64 + w] = p.h[i];
        }
    }
    copy_join();
    __syncthreads();

    // ---- scores: 16 tiles (4x4), 2/wave, K=256
    for (int ii = 0; ii < 2; ++ii) {
        int tIdx = wave * 2 + ii;
        int tm = (tIdx >> 2) * 16, tn = (tIdx & 3) * 16;
        f32x8 acc = {};
#pragma unroll
        for (int k0 = 0; k0 < 256; k0 += 32)
            acc = wmma_bf16(frag_a_rm(Ks, 256, tm, k0),
                            frag_b_nk(Qs, 256, k0, tn), acc);
#pragma unroll
        for (int r = 0; r < 8; ++r) {
            int dm, dn; cd_coord(r, dm, dn);
            S[(tm + dm) * 64 + tn + dn] = acc[r] * 0.0625f;
        }
    }
    __syncthreads();

    // ---- softmax over w (per column v); store transposed Pt[v][w]
    if (t < 64) {
        int v = t;
        float mx = -1e30f;
        for (int w = 0; w < 64; ++w) mx = fmaxf(mx, S[w * 64 + v]);
        float sum = 0.0f;
        for (int w = 0; w < 64; ++w) { float e = __expf(S[w * 64 + v] - mx); S[w * 64 + v] = e; sum += e; }
        float inv = 1.0f / sum;
        for (int w = 0; w < 64; ++w) Pt[v * 64 + w] = (bf16)(S[w * 64 + v] * inv);
    }
    __syncthreads();

    // ---- apply: Out[v][c], 4x16=64 tiles, 8/wave, K=64
    for (int ii = 0; ii < 8; ++ii) {
        int tIdx = wave * 8 + ii;
        int tm = (tIdx >> 4) * 16, tn = (tIdx & 15) * 16;
        f32x8 acc = {};
#pragma unroll
        for (int k0 = 0; k0 < 64; k0 += 32)
            acc = wmma_bf16(frag_a_rm(Pt, 64, tm, k0),
                            frag_b_nk(Vt, 64, k0, tn), acc);
#pragma unroll
        for (int r = 0; r < 8; ++r) {
            int dm, dn; cd_coord(r, dm, dn);
            int v = tm + dm, c = tn + dn;
            vcat[((size_t)(b * 64 + hh) * 64 + v) * 512 + c] = (bf16)acc[r];
        }
    }
}

// ---------------------------------------------------------------------------
// Kernel 3: width-axis attention.  One WG per (b,w).
//   T[h][g] = sum_c Qs[h][c]*Ksl[g][c] /16 ; softmax over g (per row h)
//   Out[h][c] = sum_g P[h][g]*Vt[c][g]   -> vcat[b][h][w][256:512]
// ---------------------------------------------------------------------------
__global__ __launch_bounds__(BLK)
void attn_w_kernel(const bf16* __restrict__ qT, const bf16* __restrict__ kT,
                   const bf16* __restrict__ vT, bf16* __restrict__ vcat) {
    extern __shared__ __align__(32) char smem[];
    bf16*  Qs  = (bf16*)smem;              // [64][256]  A: [h][c]
    bf16*  Ksl = Qs + 64 * 256;            // [64][256]  B(NK): [g][c]
    bf16*  Vt  = Ksl + 64 * 256;           // [256][64]  B(NK): [c][g]
    float* T   = (float*)(Vt + 256 * 64);  // [64][64]   scores [h][g]
    bf16*  P   = (bf16*)(T + 64 * 64);     // [64][64]   A: [h][g]

    const int blk = blockIdx.x;
    const int b = blk >> 6, ww = blk & 63;
    const int t = threadIdx.x, wave = t >> 5;
    const size_t base = (size_t)b * 64 * 64 * 256;   // [h][w][c] for batch b

    // ---- stage Q,K: rows at fixed w are 256 contiguous channels
#pragma unroll
    for (int j = 0; j < 8; ++j) {
        int f = t + j * BLK;               // 2048 uint4 per tensor
        int row = f >> 5, chunk = f & 31;
        size_t src = base + ((size_t)(row * 64 + ww)) * 256 + chunk * 8;
        copy16_g2l(&((uint4*)Qs)[f],  qT + src);
        copy16_g2l(&((uint4*)Ksl)[f], kT + src);
    }
    // ---- stage V transposed: [g][c] -> Vt[c][g]
#pragma unroll
    for (int j = 0; j < 8; ++j) {
        int f = t + j * BLK;
        int g = f >> 5, chunk = f & 31;
        Pack8 p;
        p.u = *(const uint4*)(vT + base + ((size_t)(g * 64 + ww)) * 256 + chunk * 8);
#pragma unroll
        for (int i = 0; i < 8; ++i)
            Vt[(chunk * 8 + i) * 64 + g] = p.h[i];
    }
    copy_join();
    __syncthreads();

    // ---- scores: 16 tiles, 2/wave, K=256
    for (int ii = 0; ii < 2; ++ii) {
        int tIdx = wave * 2 + ii;
        int tm = (tIdx >> 2) * 16, tn = (tIdx & 3) * 16;
        f32x8 acc = {};
#pragma unroll
        for (int k0 = 0; k0 < 256; k0 += 32)
            acc = wmma_bf16(frag_a_rm(Qs, 256, tm, k0),
                            frag_b_nk(Ksl, 256, k0, tn), acc);
#pragma unroll
        for (int r = 0; r < 8; ++r) {
            int dm, dn; cd_coord(r, dm, dn);
            T[(tm + dm) * 64 + tn + dn] = acc[r] * 0.0625f;
        }
    }
    __syncthreads();

    // ---- softmax over g (per row h); P[h][g]
    if (t < 64) {
        int h = t;
        float mx = -1e30f;
        for (int g = 0; g < 64; ++g) mx = fmaxf(mx, T[h * 64 + g]);
        float sum = 0.0f;
        for (int g = 0; g < 64; ++g) { float e = __expf(T[h * 64 + g] - mx); T[h * 64 + g] = e; sum += e; }
        float inv = 1.0f / sum;
        for (int g = 0; g < 64; ++g) P[h * 64 + g] = (bf16)(T[h * 64 + g] * inv);
    }
    __syncthreads();

    // ---- apply: Out[h][c], 64 tiles, 8/wave, K=64
    for (int ii = 0; ii < 8; ++ii) {
        int tIdx = wave * 8 + ii;
        int tm = (tIdx >> 4) * 16, tn = (tIdx & 15) * 16;
        f32x8 acc = {};
#pragma unroll
        for (int k0 = 0; k0 < 64; k0 += 32)
            acc = wmma_bf16(frag_a_rm(P, 64, tm, k0),
                            frag_b_nk(Vt, 64, k0, tn), acc);
#pragma unroll
        for (int r = 0; r < 8; ++r) {
            int dm, dn; cd_coord(r, dm, dn);
            int h = tm + dm, c = tn + dn;
            vcat[((size_t)(b * 64 + h) * 64 + ww) * 512 + 256 + c] = (bf16)acc[r];
        }
    }
}

// ---------------------------------------------------------------------------
// Kernel 4: 1x1 proj GEMM.  M=65536 (b,hw), K=512, N=256.
// Epilogue: +beta2, ReLU, then LDS transpose -> coalesced float4 NCHW stores.
// ---------------------------------------------------------------------------
__global__ __launch_bounds__(BLK)
void proj_kernel(const bf16* __restrict__ vcat, const bf16* __restrict__ Bproj,
                 const float* __restrict__ beta2, float* __restrict__ out) {
    extern __shared__ __align__(32) char smem[];
    bf16*  As = (bf16*)smem;                    // [128][32]
    bf16*  Bs = (bf16*)(smem + 128 * 32 * 2);   // [64][32]
    float* Ct = (float*)smem;                   // [64][128] (overlays after GEMM)

    const int mTile = blockIdx.x * 128;
    const int nTile = blockIdx.y * 64;
    const int t = threadIdx.x, wave = t >> 5;
    const int wm = (wave & 3) * 32;
    const int wn = (wave >> 2) * 32;

    f32x8 acc[2][2] = {{{}, {}}, {{}, {}}};

    for (int k0 = 0; k0 < 512; k0 += 32) {
        // A: 512 uint4, 2/thread ; B: 256 uint4, 1/thread (async DMA)
#pragma unroll
        for (int j = 0; j < 2; ++j) {
            int f = t + j * BLK;
            int row = f >> 2, q = f & 3;
            copy16_g2l(&As[row * 32 + q * 8],
                       &vcat[(size_t)(mTile + row) * 512 + k0 + q * 8]);
        }
        {
            int n = t >> 2, q = t & 3;
            copy16_g2l(&Bs[n * 32 + q * 8],
                       &Bproj[(size_t)(nTile + n) * 512 + k0 + q * 8]);
        }
        if (k0 + 32 < 512)
            __builtin_prefetch(&vcat[(size_t)(mTile + (t >> 2)) * 512 + k0 + 32], 0, 1);
        copy_join();
        __syncthreads();
#pragma unroll
        for (int i = 0; i < 2; ++i) {
            bf16x16 a = frag_a_rm(As, 32, wm + i * 16, 0);
#pragma unroll
            for (int jj = 0; jj < 2; ++jj) {
                bf16x16 b = frag_b_nk(Bs, 32, 0, wn + jj * 16);
                acc[i][jj] = wmma_bf16(a, b, acc[i][jj]);
            }
        }
        __syncthreads();
    }

    // epilogue: bias+ReLU, scatter to Ct[n][m] in LDS, then coalesced stores
#pragma unroll
    for (int i = 0; i < 2; ++i)
#pragma unroll
        for (int jj = 0; jj < 2; ++jj)
#pragma unroll
            for (int r = 0; r < 8; ++r) {
                int dm, dn; cd_coord(r, dm, dn);
                int mL = wm + i * 16 + dm;
                int nL = wn + jj * 16 + dn;
                Ct[nL * 128 + mL] = fmaxf(acc[i][jj][r] + beta2[nTile + nL], 0.0f);
            }
    __syncthreads();
    // 64 n-rows x 128 m = 2048 float4, 8/thread; NCHW rows are m-contiguous
    {
        const int b = mTile >> 12;
        const int hwBase = mTile & 4095;
#pragma unroll
        for (int j = 0; j < 8; ++j) {
            int f = t + j * BLK;
            int n = f >> 5, chunk = f & 31;
            float4 val = *(const float4*)&Ct[n * 128 + chunk * 4];
            *(float4*)&out[((size_t)b * 256 + nTile + n) * 4096 + hwBase + chunk * 4] = val;
        }
    }
}

// ---------------------------------------------------------------------------
extern "C" void kernel_launch(void* const* d_in, const int* in_sizes, int n_in,
                              void* d_out, int out_size, void* d_ws, size_t ws_size,
                              hipStream_t stream) {
    (void)in_sizes; (void)n_in; (void)out_size; (void)ws_size;
    const float* x      = (const float*)d_in[0];
    const float* w_qkv  = (const float*)d_in[1];
    const float* gamma1 = (const float*)d_in[2];
    const float* beta1  = (const float*)d_in[3];
    const float* w_proj = (const float*)d_in[4];
    const float* gamma2 = (const float*)d_in[5];
    const float* beta2  = (const float*)d_in[6];
    float* out = (float*)d_out;

    // workspace carve (~170 MB)
    char* ws = (char*)d_ws;
    bf16* Bqkv  = (bf16*)ws;  ws += (size_t)768 * 1024 * 2;
    bf16* Bproj = (bf16*)ws;  ws += (size_t)256 * 512 * 2;
    const size_t sliceElems = (size_t)16 * 64 * 64 * 256;   // per tensor
    bf16* qT   = (bf16*)ws;  ws += sliceElems * 2;
    bf16* kT   = (bf16*)ws;  ws += sliceElems * 2;
    bf16* vT   = (bf16*)ws;  ws += sliceElems * 2;
    bf16* vcat = (bf16*)ws;  ws += (size_t)16 * 4096 * 512 * 2;

    {   // 0: fold BN into weights
        int total = 768 * 1024 + 256 * 512;
        prep_weights_kernel<<<(total + BLK - 1) / BLK, BLK, 0, stream>>>(
            w_qkv, gamma1, w_proj, gamma2, Bqkv, Bproj);
    }
    {   // 1: patchify conv GEMM (512 x 12 tiles of 128x64)
        size_t sm = (128 * 32 + 64 * 32) * sizeof(bf16);
        conv_qkv_kernel<<<dim3(512, 12), BLK, sm, stream>>>(
            x, Bqkv, beta1, qT, kT, vT);
    }
    {   // 2/3: axial attention, one WG per (b,h) / (b,w); ~120KB LDS each
        size_t sm = (size_t)(64 * 256 * 2 + 256 * 64) * sizeof(bf16)
                  + 64 * 64 * sizeof(float) + 64 * 64 * sizeof(bf16);
        attn_h_kernel<<<16 * 64, BLK, sm, stream>>>(qT, kT, vT, vcat);
        attn_w_kernel<<<16 * 64, BLK, sm, stream>>>(qT, kT, vT, vcat);
    }
    {   // 4: proj GEMM + BN + ReLU (512 x 4 tiles), 32KB LDS (Ct epilogue)
        size_t sm = 64 * 128 * sizeof(float);
        proj_kernel<<<dim3(512, 4), BLK, sm, stream>>>(vcat, Bproj, beta2, out);
    }
}
